// DGMCellLSTM_83915071029863
// MI455X (gfx1250) — matrix operand order repaired and checked by reading, not compiled, not run on hardware
//
#include <hip/hip_runtime.h>

#define B_ROWS 262144
#define D_IN   8
#define H      128
#define NL     3
#define TILES  8
#define ROWS_PER_BLOCK (TILES * 16)
#define TPB    256

// Padded LDS row strides (in halves) -> 272B / 80B rows: 16B-aligned, bank-conflict-free fragments
#define SWS  136   // W tile row stride (H + 8)
#define SUS  40    // U tile row stride (32 + 8)
#define ACTS 136   // activation tile row stride

// Pre-converted f16 weight-image geometry in d_ws (units: halves)
#define WSLAB      (4 * H * SWS)                    // one layer's 4 gate W tiles
#define NW_HALVES  (NL * WSLAB)
#define USLAB      (4 * H * SUS)                    // one layer's 4 gate U tiles (K-padded)
#define NU_HALVES  (NL * USLAB)
#define NSW_HALVES (H * SUS)                        // init Sw tile (K-padded)
#define WS_HALVES  (NW_HALVES + NU_HALVES + NSW_HALVES)
#define WS_BYTES   (WS_HALVES * 2)

#define HAS_ASYNC_LDS __has_builtin(__builtin_amdgcn_global_load_async_to_lds_b128)

typedef __attribute__((ext_vector_type(16))) _Float16 v16h;
typedef __attribute__((ext_vector_type(8)))  _Float16 v8h;
typedef __attribute__((ext_vector_type(8)))  float    v8f;
typedef __attribute__((ext_vector_type(4)))  float    v4f;
typedef __attribute__((ext_vector_type(4)))  int      v4i;

__device__ __forceinline__ float sigf(float x) { return 1.0f / (1.0f + __expf(-x)); }
__device__ __forceinline__ float tanhfast(float x) { return 2.0f * sigf(2.0f * x) - 1.0f; }

__device__ __forceinline__ v16h pack16(v8h a, v8h b) {
    v16h r;
#pragma unroll
    for (int i = 0; i < 8; i++) { r[i] = a[i]; r[i + 8] = b[i]; }
    return r;
}

// A fragment (16x32 f16) from row-major [16][ACTS] f16 LDS tile.
__device__ __forceinline__ v16h ldA(const _Float16* tile, int kbase, int lane) {
    int m  = lane & 15;
    int sh = (lane & 16) ? 8 : 0;
    const _Float16* p = tile + m * ACTS + kbase + sh;
    return pack16(*(const v8h*)p, *(const v8h*)(p + 16));
}

// B fragment (32x16 f16) from [N][stride] f16 LDS matrix (weights stored transposed).
__device__ __forceinline__ v16h ldB(const _Float16* mat, int ntile, int kbase, int stride, int lane) {
    int n  = lane & 15;
    int sh = (lane & 16) ? 8 : 0;
    const _Float16* p = mat + (ntile * 16 + n) * stride + kbase + sh;
    return pack16(*(const v8h*)p, *(const v8h*)(p + 16));
}

__device__ __forceinline__ v8f wmma_acc(v16h a, v16h b, v8f c) {
    return __builtin_amdgcn_wmma_f32_16x16x32_f16(false, a, false, b, (short)0, c, false, false);
}

// acc[16x16] = x_pad[16x32] @ U_pad[32x16] + act[16x128] @ W[128x16]; two chains for WMMA ILP
__device__ __forceinline__ v8f gate_gemm(v16h xfrag, const _Float16* sUg, const _Float16* sWg,
                                         const _Float16* act, int ntile, int lane) {
    v8f a0 = {}, a1 = {};
    a0 = wmma_acc(xfrag, ldB(sUg, ntile, 0, SUS, lane), a0);
    a1 = wmma_acc(ldA(act, 0,  lane), ldB(sWg, ntile, 0,  SWS, lane), a1);
    a0 = wmma_acc(ldA(act, 32, lane), ldB(sWg, ntile, 32, SWS, lane), a0);
    a1 = wmma_acc(ldA(act, 64, lane), ldB(sWg, ntile, 64, SWS, lane), a1);
    a0 = wmma_acc(ldA(act, 96, lane), ldB(sWg, ntile, 96, SWS, lane), a0);
#pragma unroll
    for (int i = 0; i < 8; i++) a0[i] += a1[i];
    return a0;
}

#if HAS_ASYNC_LDS
__device__ __forceinline__ void async_cp16(const void* g, void* l) {
    __builtin_amdgcn_global_load_async_to_lds_b128(
        (__attribute__((address_space(1))) v4i*)(void*)g,
        (__attribute__((address_space(3))) v4i*)l, 0, 0);
}
__device__ __forceinline__ void async_wait0() {
#if __has_builtin(__builtin_amdgcn_s_wait_asynccnt)
    __builtin_amdgcn_s_wait_asynccnt(0);
#else
    asm volatile("s_wait_asynccnt 0x0" ::: "memory");
#endif
}
// contiguous global->LDS async copy, 16B per lane per issue
__device__ __forceinline__ void stage_async(const void* g, void* l, int bytes, int tid) {
    const char* gs = (const char*)g;
    char* ls = (char*)l;
    for (int o = tid * 16; o < bytes; o += TPB * 16)
        async_cp16(gs + o, ls + o);
}

// One-time: convert all weights f32 -> f16 into d_ws, already in LDS-image layout
__global__ __launch_bounds__(TPB)
void dgm_prep(const float* __restrict__ Sw_w,
              const float* __restrict__ Uz, const float* __restrict__ Ug,
              const float* __restrict__ Ur, const float* __restrict__ Uh,
              const float* __restrict__ Wz, const float* __restrict__ Wg,
              const float* __restrict__ Wr, const float* __restrict__ Wh,
              _Float16* __restrict__ ws) {
    int e = blockIdx.x * TPB + threadIdx.x;
    if (e >= WS_HALVES) return;
    float v = 0.0f;
    if (e < NW_HALVES) {
        int l = e / WSLAB, rem = e % WSLAB;
        int g = rem / (H * SWS), rem2 = rem % (H * SWS);
        int n = rem2 / SWS, k = rem2 % SWS;
        const float* Wp[4] = {Wz, Wg, Wr, Wh};
        if (k < H) v = Wp[g][l * H * H + k * H + n];
    } else if (e < NW_HALVES + NU_HALVES) {
        int e2 = e - NW_HALVES;
        int l = e2 / USLAB, rem = e2 % USLAB;
        int g = rem / (H * SUS), rem2 = rem % (H * SUS);
        int n = rem2 / SUS, k = rem2 % SUS;
        const float* Up[4] = {Uz, Ug, Ur, Uh};
        if (k < D_IN) v = Up[g][l * D_IN * H + k * H + n];
    } else {
        int e2 = e - NW_HALVES - NU_HALVES;
        int n = e2 / SUS, k = e2 % SUS;
        if (k < D_IN) v = Sw_w[k * H + n];
    }
    ws[e] = (_Float16)v;
}
#endif  // HAS_ASYNC_LDS

template <bool USE_ASYNC>
__global__ __launch_bounds__(TPB)
void dgm_kernel(const float* __restrict__ x,
                const float* __restrict__ Sw_w, const float* __restrict__ Sw_b,
                const float* __restrict__ Uz, const float* __restrict__ Wz, const float* __restrict__ bz,
                const float* __restrict__ Ug, const float* __restrict__ Wg, const float* __restrict__ bg,
                const float* __restrict__ Ur, const float* __restrict__ Wr, const float* __restrict__ br,
                const float* __restrict__ Uh, const float* __restrict__ Wh, const float* __restrict__ bh,
                const float* __restrict__ Wf_w, const float* __restrict__ Wf_b,
                const _Float16* __restrict__ wimg,
                float* __restrict__ out) {
    __shared__ alignas(16) _Float16 sW[4][H * SWS];     // gate weights [N][K], padded (136 KB)
    __shared__ alignas(16) _Float16 sU[4][H * SUS];     // U weights [N][32pad] (40 KB); sU[0] = Sw in init
    __shared__ alignas(16) float    sS[TILES][16 * H];  // S state per row-tile (64 KB)
    __shared__ alignas(16) float    sX[TILES][16 * D_IN];
    __shared__ alignas(16) _Float16 sAct[16 * ACTS];    // sigmoid(S) f16 (A operand)
    __shared__ alignas(16) _Float16 sSR[16 * ACTS];     // sigmoid(S)*R f16 (A operand for H gate)

    const int tid  = threadIdx.x;
    const int lane = tid & 31;
    const int wave = tid >> 5;                       // ntile 0..7 (column block)
    const long rowBase = (long)blockIdx.x * ROWS_PER_BLOCK;

    const int m0  = (lane & 16) ? 8 : 0;             // D-layout row base for this lane
    const int col = wave * 16 + (lane & 15);         // D-layout column for this lane

    // ---- stage x tiles + init weights (Sw K-padded, into sU[0]) ----
    if constexpr (USE_ASYNC) {
#if HAS_ASYNC_LDS
        stage_async(wimg + NW_HALVES + NU_HALVES, &sU[0][0], NSW_HALVES * 2, tid);
        stage_async(x + rowBase * D_IN, &sX[0][0], ROWS_PER_BLOCK * D_IN * 4, tid);
        async_wait0();
#endif
    } else {
        for (int e = tid; e < H * SUS; e += TPB) {
            int n = e / SUS, k = e % SUS;
            sU[0][e] = (_Float16)(k < D_IN ? Sw_w[k * H + n] : 0.0f);
        }
        for (int e = tid; e < TILES * 16 * D_IN; e += TPB) {
            int t = e / (16 * D_IN);
            int r = e % (16 * D_IN);
            sX[t][r] = x[(rowBase + t * 16) * D_IN + r];
        }
    }
    __syncthreads();

    // ---- init: S = x @ Sw_w + Sw_b ----
    for (int t = 0; t < TILES; t++) {
        v16h xf;
#pragma unroll
        for (int i = 0; i < 16; i++) xf[i] = (_Float16)0.0f;
        if (lane < 16) {
#pragma unroll
            for (int k = 0; k < D_IN; k++) xf[k] = (_Float16)sX[t][lane * D_IN + k];
        }
        v8f acc = {};
        acc = wmma_acc(xf, ldB(sU[0], wave, 0, SUS, lane), acc);
        float bias = Sw_b[col];
#pragma unroll
        for (int r = 0; r < 8; r++) sS[t][(m0 + r) * H + col] = acc[r] + bias;
    }
    __syncthreads();

    for (int layer = 0; layer < NL; layer++) {
        const float* bgate[4] = {bz + layer * H, bg + layer * H, br + layer * H, bh + layer * H};

        // ---- stage this layer's W + U tiles into LDS ----
        if constexpr (USE_ASYNC) {
#if HAS_ASYNC_LDS
            stage_async(wimg + (size_t)layer * WSLAB, &sW[0][0], WSLAB * 2, tid);
            stage_async(wimg + NW_HALVES + (size_t)layer * USLAB, &sU[0][0], USLAB * 2, tid);
            async_wait0();
#endif
        } else {
            const float* Wgate[4] = {Wz + layer * H * H, Wg + layer * H * H,
                                     Wr + layer * H * H, Wh + layer * H * H};
            const float* Ugate[4] = {Uz + layer * D_IN * H, Ug + layer * D_IN * H,
                                     Ur + layer * D_IN * H, Uh + layer * D_IN * H};
            for (int e = tid; e < 4 * (H * H / 4); e += TPB) {
                int g   = e >> 12;
                int rem = e & 4095;
                int k   = rem >> 5;
                int n0  = (rem & 31) * 4;
                v4f w4 = *(const v4f*)(Wgate[g] + k * H + n0);
#pragma unroll
                for (int j = 0; j < 4; j++)
                    sW[g][(n0 + j) * SWS + k] = (_Float16)w4[j];
            }
            for (int e = tid; e < 4 * H * SUS; e += TPB) {
                int g = e / (H * SUS), rem = e % (H * SUS);
                int n = rem / SUS, k = rem % SUS;
                sU[g][rem] = (_Float16)(k < D_IN ? Ugate[g][k * H + n] : 0.0f);
            }
        }
        __syncthreads();

        for (int t = 0; t < TILES; t++) {
            // sigmoid(S) -> f16 A-operand tile (padded stride)
            for (int e = tid; e < 16 * H; e += TPB) {
                int m = e >> 7, c = e & 127;
                sAct[m * ACTS + c] = (_Float16)sigf(sS[t][e]);
            }
            __syncthreads();

            v16h xf;
#pragma unroll
            for (int i = 0; i < 16; i++) xf[i] = (_Float16)0.0f;
            if (lane < 16) {
#pragma unroll
                for (int k = 0; k < D_IN; k++) xf[k] = (_Float16)sX[t][lane * D_IN + k];
            }

            float sa[8];
#pragma unroll
            for (int r = 0; r < 8; r++) sa[r] = (float)sAct[(m0 + r) * ACTS + col];

            // R gate, then write S*R tile
            v8f aR = gate_gemm(xf, sU[2], sW[2], sAct, wave, lane);
            float bR = bgate[2][col];
#pragma unroll
            for (int r = 0; r < 8; r++) {
                float Rv = sigf(aR[r] + bR);
                sSR[(m0 + r) * ACTS + col] = (_Float16)(sa[r] * Rv);
            }

            // Z and G gates overlap the SR stores; barrier before H reads sSR
            v8f aZ = gate_gemm(xf, sU[0], sW[0], sAct, wave, lane);
            v8f aG = gate_gemm(xf, sU[1], sW[1], sAct, wave, lane);
            __syncthreads();
            v8f aH = gate_gemm(xf, sU[3], sW[3], sSR, wave, lane);

            float bZ = bgate[0][col], bG = bgate[1][col], bH = bgate[3][col];
#pragma unroll
            for (int r = 0; r < 8; r++) {
                float Zv = sigf(aZ[r] + bZ);
                float Gv = sigf(aG[r] + bG);
                float Hv = tanhfast(aH[r] + bH);
                sS[t][(m0 + r) * H + col] = (1.0f - Gv) * Hv + Zv * sa[r];
            }
            __syncthreads();
        }
    }

    // ---- out = S @ Wf_w + Wf_b  (one tile per wave, 16 lanes per tile) ----
    {
        float wfb = Wf_b[0];
        int t = wave;                 // TILES == 8 waves
        if (lane < 16) {
            float accum = 0.0f;
            for (int c = 0; c < H; c++) accum += sS[t][lane * H + c] * Wf_w[c];
            out[rowBase + t * 16 + lane] = accum + wfb;
        }
    }
}

extern "C" void kernel_launch(void* const* d_in, const int* in_sizes, int n_in,
                              void* d_out, int out_size, void* d_ws, size_t ws_size,
                              hipStream_t stream) {
    const float* x    = (const float*)d_in[0];
    const float* Sw_w = (const float*)d_in[1];
    const float* Sw_b = (const float*)d_in[2];
    const float* Uz   = (const float*)d_in[3];
    const float* Wz   = (const float*)d_in[4];
    const float* bz   = (const float*)d_in[5];
    const float* Ug   = (const float*)d_in[6];
    const float* Wg   = (const float*)d_in[7];
    const float* bg   = (const float*)d_in[8];
    const float* Ur   = (const float*)d_in[9];
    const float* Wr   = (const float*)d_in[10];
    const float* br   = (const float*)d_in[11];
    const float* Uh   = (const float*)d_in[12];
    const float* Wh   = (const float*)d_in[13];
    const float* bh   = (const float*)d_in[14];
    const float* Wf_w = (const float*)d_in[15];
    const float* Wf_b = (const float*)d_in[16];
    float* out = (float*)d_out;

    dim3 grid(B_ROWS / ROWS_PER_BLOCK);
    dim3 block(TPB);

#if HAS_ASYNC_LDS
    if (ws_size >= (size_t)WS_BYTES) {
        _Float16* wimg = (_Float16*)d_ws;
        dgm_prep<<<dim3((WS_HALVES + TPB - 1) / TPB), block, 0, stream>>>(
            Sw_w, Uz, Ug, Ur, Uh, Wz, Wg, Wr, Wh, wimg);
        dgm_kernel<true><<<grid, block, 0, stream>>>(
            x, Sw_w, Sw_b, Uz, Wz, bz, Ug, Wg, bg, Ur, Wr, br, Uh, Wh, bh,
            Wf_w, Wf_b, wimg, out);
        return;
    }
#endif
    dgm_kernel<false><<<grid, block, 0, stream>>>(
        x, Sw_w, Sw_b, Uz, Wz, bz, Ug, Wg, bg, Ur, Wr, br, Uh, Wh, bh,
        Wf_w, Wf_b, nullptr, out);
}
